// GRUEncoder_54589034332754
// MI455X (gfx1250) — compile-verified, matrix operand order checked by reading, not run on hardware
//
#include <hip/hip_runtime.h>
#include <hip/hip_bf16.h>

typedef __attribute__((ext_vector_type(16))) _Float16 v16h;
typedef __attribute__((ext_vector_type(8)))  _Float16 v8h;
typedef __attribute__((ext_vector_type(8)))  float    v8f;

#define TT     8192
#define HID    512
#define H3     1536
#define DMODEL 1024
#define INPD   768

union frag16 { v16h v; v8h h8[2]; };

#define WMMA_F16(A, B, Cc) \
    __builtin_amdgcn_wmma_f32_16x16x32_f16(false, (A), false, (B), (short)0, (Cc), false, false)

// ---------------------------------------------------------------------------
// f32 -> f16 conversion (grid-stride)
// ---------------------------------------------------------------------------
__global__ void cvt_f32_f16(const float* __restrict__ s, _Float16* __restrict__ d, size_t n) {
    size_t i = blockIdx.x * (size_t)blockDim.x + threadIdx.x;
    size_t stride = gridDim.x * (size_t)blockDim.x;
    for (; i < n; i += stride) d[i] = (_Float16)s[i];
}

// ---------------------------------------------------------------------------
// C[M,N] f32 = A16[M,lda] @ W16[N,ldw]^T + bias[N]
// Each wave computes a 32x64 strip: 2 M-tiles x 4 N-tiles = 8 accumulators.
// Every B fragment feeds 2 WMMAs, every A fragment feeds 4. K-loop is
// double-buffered in 64-wide super-chunks (K % 64 == 0) so one stage's loads
// fly while the other stage's 8 WMMAs execute.
// A-frag layout (CDNA5 16-bit A): lane L -> row M=(L&15), koff=(L>>4)*8.
// C/D layout: c[r] at lane L -> M = r + 8*(L>>4), N = (L&15).
// ---------------------------------------------------------------------------
__global__ void wmma_gemm_bias(const _Float16* __restrict__ A, int lda,
                               const _Float16* __restrict__ W, int ldw,
                               const float* __restrict__ bias,
                               float* __restrict__ C, int ldc, int K)
{
    const int lane    = threadIdx.x & 31;
    const int wave    = threadIdx.x >> 5;
    const int n0      = (blockIdx.x * 8 + wave) * 64;   // 4 N-tiles per wave
    const int m0      = blockIdx.y * 32;                // 2 M-tiles per wave
    const int halfSel = lane >> 4;
    const int koff    = halfSel * 8;
    const int l15     = lane & 15;

    const _Float16* apr0 = A + (size_t)(m0 + l15)      * lda + koff;
    const _Float16* apr1 = A + (size_t)(m0 + 16 + l15) * lda + koff;
    const _Float16* wp0  = W + (size_t)(n0 + l15)      * ldw + koff;
    const _Float16* wp1  = W + (size_t)(n0 + 16 + l15) * ldw + koff;
    const _Float16* wp2  = W + (size_t)(n0 + 32 + l15) * ldw + koff;
    const _Float16* wp3  = W + (size_t)(n0 + 48 + l15) * ldw + koff;

    v8f c00 = {}, c01 = {}, c02 = {}, c03 = {};
    v8f c10 = {}, c11 = {}, c12 = {}, c13 = {};
    frag16 af[2][2];     // [stage][m-tile]
    frag16 bf[2][4];     // [stage][n-tile]

#define LOAD_STAGE(st, kk)                                                                      \
    af[st][0].h8[0] = *(const v8h*)(apr0 + (kk)); af[st][0].h8[1] = *(const v8h*)(apr0 + (kk) + 16); \
    af[st][1].h8[0] = *(const v8h*)(apr1 + (kk)); af[st][1].h8[1] = *(const v8h*)(apr1 + (kk) + 16); \
    bf[st][0].h8[0] = *(const v8h*)(wp0 + (kk));  bf[st][0].h8[1] = *(const v8h*)(wp0 + (kk) + 16);  \
    bf[st][1].h8[0] = *(const v8h*)(wp1 + (kk));  bf[st][1].h8[1] = *(const v8h*)(wp1 + (kk) + 16);  \
    bf[st][2].h8[0] = *(const v8h*)(wp2 + (kk));  bf[st][2].h8[1] = *(const v8h*)(wp2 + (kk) + 16);  \
    bf[st][3].h8[0] = *(const v8h*)(wp3 + (kk));  bf[st][3].h8[1] = *(const v8h*)(wp3 + (kk) + 16);

#define MMA_STAGE(st)                                        \
    c00 = WMMA_F16(af[st][0].v, bf[st][0].v, c00);           \
    c10 = WMMA_F16(af[st][1].v, bf[st][0].v, c10);           \
    c01 = WMMA_F16(af[st][0].v, bf[st][1].v, c01);           \
    c11 = WMMA_F16(af[st][1].v, bf[st][1].v, c11);           \
    c02 = WMMA_F16(af[st][0].v, bf[st][2].v, c02);           \
    c12 = WMMA_F16(af[st][1].v, bf[st][2].v, c12);           \
    c03 = WMMA_F16(af[st][0].v, bf[st][3].v, c03);           \
    c13 = WMMA_F16(af[st][1].v, bf[st][3].v, c13);

    LOAD_STAGE(0, 0)
    for (int k = 0; k < K; k += 64) {
        LOAD_STAGE(1, k + 32)      // stage-1 loads fly during stage-0 WMMAs
        MMA_STAGE(0)
        const int k2 = k + 64;
        if (k2 < K) { LOAD_STAGE(0, k2) }   // stage-0 loads fly during stage-1 WMMAs
        MMA_STAGE(1)
    }
#undef LOAD_STAGE
#undef MMA_STAGE

    const float bn0 = bias[n0 + l15];
    const float bn1 = bias[n0 + 16 + l15];
    const float bn2 = bias[n0 + 32 + l15];
    const float bn3 = bias[n0 + 48 + l15];
#pragma unroll
    for (int r = 0; r < 8; ++r) {
        const size_t mA = (size_t)(m0 + r + halfSel * 8) * ldc;
        const size_t mB = (size_t)(m0 + 16 + r + halfSel * 8) * ldc;
        C[mA + n0 + l15]      = c00[r] + bn0;
        C[mA + n0 + 16 + l15] = c01[r] + bn1;
        C[mA + n0 + 32 + l15] = c02[r] + bn2;
        C[mA + n0 + 48 + l15] = c03[r] + bn3;
        C[mB + n0 + l15]      = c10[r] + bn0;
        C[mB + n0 + 16 + l15] = c11[r] + bn1;
        C[mB + n0 + 32 + l15] = c12[r] + bn2;
        C[mB + n0 + 48 + l15] = c13[r] + bn3;
    }
}

// ---------------------------------------------------------------------------
// Persistent GRU scan: one block per direction (grid.x = 2), 512 threads =
// 16 waves. Per step: gh = h @ Whh^T via WMMA (h broadcast into all 16 A
// rows from LDS), then f32 gate math. kc-outer loop: one LDS a-fragment
// feeds 6 WMMAs (6 N-tiles / wave, 16 waves x 96 outputs = 1536).
// ---------------------------------------------------------------------------
__global__ void gru_scan(const float* __restrict__ gx_base,   // 2 x T x 1536
                         const _Float16* __restrict__ Whh16,  // 2 x 1536 x 512
                         const float* __restrict__ bhh,       // 2 x 1536
                         _Float16* __restrict__ yout,         // T x 1024 (dir offset applied)
                         float* __restrict__ yf32,            // T x 1024 or null
                         float* __restrict__ hT,              // 2 x 512
                         int T)
{
    const int dir = blockIdx.x;
    const float*    gx = gx_base + (size_t)dir * T * H3;
    const _Float16* Wh = Whh16 + (size_t)dir * H3 * HID;
    const float*    bh = bhh + dir * H3;
    _Float16*      y16 = yout + dir * HID;
    float*          yf = yf32 ? yf32 + dir * HID : nullptr;

    __shared__ __align__(16) _Float16 h16[HID];
    __shared__ float h32[HID];
    __shared__ float gh[H3];
    __shared__ float bh_s[H3];

    const int tid  = threadIdx.x;          // 0..511
    const int lane = tid & 31;
    const int wave = tid >> 5;             // 0..15
    const int koff = (lane >> 4) * 8;
    const int l15  = lane & 15;

    h16[tid] = (_Float16)0.0f;
    h32[tid] = 0.0f;
    bh_s[tid]           = bh[tid];
    bh_s[HID + tid]     = bh[HID + tid];
    bh_s[2 * HID + tid] = bh[2 * HID + tid];

    // Loop-invariant Whh row pointers for this wave's 6 N-tiles.
    const _Float16* wrow[6];
#pragma unroll
    for (int i = 0; i < 6; ++i)
        wrow[i] = Wh + (size_t)((wave * 6 + i) * 16 + l15) * HID + koff;

    for (int s = 0; s < T; ++s) {
        const int t = dir ? (T - 1 - s) : s;
        __syncthreads();                   // h16 from previous step visible

        // ---- gh = h @ Whh^T ----
        v8f acc[6] = {};
        for (int kc = 0; kc < HID; kc += 32) {
            frag16 a;
            const _Float16* hp = h16 + kc + koff;   // LDS broadcast
            a.h8[0] = *(const v8h*)(hp);
            a.h8[1] = *(const v8h*)(hp + 16);
#pragma unroll
            for (int i = 0; i < 6; ++i) {
                frag16 b;
                b.h8[0] = *(const v8h*)(wrow[i] + kc);
                b.h8[1] = *(const v8h*)(wrow[i] + kc + 16);
                acc[i] = WMMA_F16(a.v, b.v, acc[i]);
            }
        }
        if (lane < 16) {
#pragma unroll
            for (int i = 0; i < 6; ++i)
                gh[(wave * 6 + i) * 16 + lane] = acc[i][0];   // row M=0 == h row
        }
        __syncthreads();                   // gh complete

        // ---- gates (f32) ----
        {
            const float* gxt = gx + (size_t)t * H3;
            if (s + 1 < T) {               // pull next step's gx row closer
                const float* gxn = gx + (size_t)(dir ? t - 1 : t + 1) * H3;
                __builtin_prefetch(gxn + tid, 0, 2);
                __builtin_prefetch(gxn + HID + tid, 0, 2);
                __builtin_prefetch(gxn + 2 * HID + tid, 0, 2);
            }
            const float ghr = gh[tid]           + bh_s[tid];
            const float ghz = gh[HID + tid]     + bh_s[HID + tid];
            const float ghn = gh[2 * HID + tid] + bh_s[2 * HID + tid];
            const float r = 1.0f / (1.0f + __expf(-(gxt[tid] + ghr)));
            const float z = 1.0f / (1.0f + __expf(-(gxt[HID + tid] + ghz)));
            const float n = tanhf(gxt[2 * HID + tid] + r * ghn);
            const float h2 = (1.0f - z) * n + z * h32[tid];
            h32[tid] = h2;
            h16[tid] = (_Float16)h2;
            y16[(size_t)t * DMODEL + tid] = (_Float16)h2;
            if (yf) yf[(size_t)t * DMODEL + tid] = h2;
        }
    }
    __syncthreads();
    hT[dir * HID + tid] = h32[tid];
}

// ---------------------------------------------------------------------------
// out[t,:] += layernorm(y[t,:]) * g + b     (one block per row, D = 1024)
// ---------------------------------------------------------------------------
__device__ __forceinline__ float block_reduce_sum(float v, float* red) {
    red[threadIdx.x] = v; __syncthreads();
    for (int st = 128; st > 0; st >>= 1) {
        if (threadIdx.x < st) red[threadIdx.x] += red[threadIdx.x + st];
        __syncthreads();
    }
    float r = red[0];
    __syncthreads();
    return r;
}

__global__ void ln_add(const float* __restrict__ y, const float* __restrict__ g,
                       const float* __restrict__ b, float* __restrict__ out, int D)
{
    __shared__ float red[256];
    const size_t t = blockIdx.x;
    const float* row = y + t * D;
    float s = 0.0f;
    for (int j = threadIdx.x; j < D; j += 256) s += row[j];
    const float mean = block_reduce_sum(s, red) / D;
    float v = 0.0f;
    for (int j = threadIdx.x; j < D; j += 256) { float d = row[j] - mean; v += d * d; }
    const float rstd = rsqrtf(block_reduce_sum(v, red) / D + 1e-5f);
    for (int j = threadIdx.x; j < D; j += 256)
        out[t * D + j] += (row[j] - mean) * rstd * g[j] + b[j];
}

// ---------------------------------------------------------------------------
// hidden head: LN(hcat[l,:1024]) @ fc_W^T + fc_b  -> outh[l, :512]
// ---------------------------------------------------------------------------
__global__ void hidden_head(const float* __restrict__ hcat,  // 5 x 1024
                            const float* __restrict__ g, const float* __restrict__ b,
                            const float* __restrict__ fcW,   // 512 x 1024
                            const float* __restrict__ fcb,
                            float* __restrict__ outh)        // 5 x 512
{
    __shared__ float row[DMODEL];
    __shared__ float red[256];
    const int l = blockIdx.x;
    const float* src = hcat + (size_t)l * DMODEL;
    for (int j = threadIdx.x; j < DMODEL; j += 256) row[j] = src[j];
    __syncthreads();
    float s = 0.0f;
    for (int j = threadIdx.x; j < DMODEL; j += 256) s += row[j];
    const float mean = block_reduce_sum(s, red) / DMODEL;
    float v = 0.0f;
    for (int j = threadIdx.x; j < DMODEL; j += 256) { float d = row[j] - mean; v += d * d; }
    const float rstd = rsqrtf(block_reduce_sum(v, red) / DMODEL + 1e-5f);
    for (int j = threadIdx.x; j < DMODEL; j += 256)
        row[j] = (row[j] - mean) * rstd * g[j] + b[j];
    __syncthreads();
    for (int j = threadIdx.x; j < HID; j += 256) {
        float acc = fcb[j];
        const float* w = fcW + (size_t)j * DMODEL;
        for (int k = 0; k < DMODEL; ++k) acc += row[k] * w[k];
        outh[(size_t)l * HID + j] = acc;
    }
}

// ---------------------------------------------------------------------------
// Host orchestration
// ---------------------------------------------------------------------------
extern "C" void kernel_launch(void* const* d_in, const int* in_sizes, int n_in,
                              void* d_out, int out_size, void* d_ws, size_t ws_size,
                              hipStream_t stream) {
    const float* x      = (const float*)d_in[0];
    const float* Wih0   = (const float*)d_in[1];
    const float* Whh0   = (const float*)d_in[2];
    const float* bih0   = (const float*)d_in[3];
    const float* bhh0   = (const float*)d_in[4];
    const float* Wih    = (const float*)d_in[5];
    const float* Whh    = (const float*)d_in[6];
    const float* bih    = (const float*)d_in[7];
    const float* bhh    = (const float*)d_in[8];
    const float* lnh_g  = (const float*)d_in[9];
    const float* lnh_b  = (const float*)d_in[10];
    const float* lne_g  = (const float*)d_in[11];
    const float* lne_b  = (const float*)d_in[12];
    const float* fc_W   = (const float*)d_in[13];
    const float* fc_b   = (const float*)d_in[14];
    const float* skip_W = (const float*)d_in[15];
    const float* skip_b = (const float*)d_in[16];

    float* out    = (float*)d_out;                       // T x 1024
    float* out_h  = (float*)d_out + (size_t)TT * DMODEL; // 5 x 512

    // Workspace carve-up (256B aligned)
    char* ws = (char*)d_ws;
    size_t off = 0;
    auto take = [&](size_t bytes) -> void* {
        off = (off + 255) & ~(size_t)255;
        void* p = ws + off;
        off += bytes;
        return p;
    };
    _Float16* x16     = (_Float16*)take((size_t)TT * INPD * 2);
    _Float16* skipW16 = (_Float16*)take((size_t)DMODEL * INPD * 2);
    _Float16* buf0    = (_Float16*)take((size_t)TT * DMODEL * 2);
    _Float16* buf1    = (_Float16*)take((size_t)TT * DMODEL * 2);
    float*    gx      = (float*)   take((size_t)2 * TT * H3 * 4);
    _Float16* Wih16   = (_Float16*)take((size_t)2 * H3 * DMODEL * 2);
    _Float16* Whh16   = (_Float16*)take((size_t)2 * H3 * HID * 2);
    float*    yfin    = (float*)   take((size_t)TT * DMODEL * 4);
    float*    hT      = (float*)   take((size_t)5 * 2 * HID * 4);

    // Convert inputs used by WMMA GEMMs to f16
    cvt_f32_f16<<<1024, 256, 0, stream>>>(x, x16, (size_t)TT * INPD);
    cvt_f32_f16<<<512, 256, 0, stream>>>(skip_W, skipW16, (size_t)DMODEL * INPD);

    const _Float16* curA = x16;

    for (int l = 0; l < 5; ++l) {
        const float* Wi = (l == 0) ? Wih0 : Wih + (size_t)(l - 1) * 2 * H3 * (2 * HID);
        const float* Wh = (l == 0) ? Whh0 : Whh + (size_t)(l - 1) * 2 * H3 * HID;
        const float* bi = (l == 0) ? bih0 : bih + (size_t)(l - 1) * 2 * H3;
        const float* bh = (l == 0) ? bhh0 : bhh + (size_t)(l - 1) * 2 * H3;
        const int K = (l == 0) ? INPD : DMODEL;

        cvt_f32_f16<<<1024, 256, 0, stream>>>(Wi, Wih16, (size_t)2 * H3 * K);
        cvt_f32_f16<<<1024, 256, 0, stream>>>(Wh, Whh16, (size_t)2 * H3 * HID);

        // gx[dir] = curA @ Wih[dir]^T + bih[dir]   (T x 1536)
        for (int dir = 0; dir < 2; ++dir) {
            dim3 grid(H3 / 512, TT / 32);     // 8 waves x 64 cols; 32 rows/block
            wmma_gemm_bias<<<grid, 256, 0, stream>>>(
                curA, K,
                Wih16 + (size_t)dir * H3 * K, K,
                bi + (size_t)dir * H3,
                gx + (size_t)dir * TT * H3, H3, K);
        }

        _Float16* nextA = (l & 1) ? buf1 : buf0;
        float* yf = (l == 4) ? yfin : nullptr;
        gru_scan<<<2, 512, 0, stream>>>(gx, Whh16, bh, nextA, yf,
                                        hT + (size_t)l * 2 * HID, TT);
        curA = nextA;
    }

    // out = x @ skip_W^T + skip_b
    {
        dim3 grid(DMODEL / 512, TT / 32);
        wmma_gemm_bias<<<grid, 256, 0, stream>>>(x16, INPD, skipW16, INPD,
                                                 skip_b, out, DMODEL, INPD);
    }
    // out += layernorm(y_last)
    ln_add<<<TT, 256, 0, stream>>>(yfin, lne_g, lne_b, out, DMODEL);

    // hidden head: LN(concat(h_fwd, h_bwd)) @ fc_W^T + fc_b
    hidden_head<<<5, 256, 0, stream>>>(hT, lnh_g, lnh_b, fc_W, fc_b, out_h);

    (void)in_sizes; (void)n_in; (void)out_size; (void)ws_size;
}
